// SelfAttention_75531294867929
// MI455X (gfx1250) — matrix operand order repaired and checked by reading, not compiled, not run on hardware
//
#include <hip/hip_runtime.h>

#define N_POS 4096
#define C_CH  64
#define CKQ   8

typedef _Float16 v16h __attribute__((ext_vector_type(16)));
typedef _Float16 v8h  __attribute__((ext_vector_type(8)));
typedef _Float16 h2   __attribute__((ext_vector_type(2)));
typedef float    v8f  __attribute__((ext_vector_type(8)));
typedef unsigned int u32x4 __attribute__((ext_vector_type(4)));
typedef int          i32x4 __attribute__((ext_vector_type(4)));
typedef int          i32x8 __attribute__((ext_vector_type(8)));

// ---------------------------------------------------------------------------
// TDM: 2D f32 tile  global -> LDS (D# group0/group1 per CDNA5 ISA §8).
// ---------------------------------------------------------------------------
__device__ __forceinline__ void tdm_load_2d_f32(unsigned lds_off, const void* gaddr,
                                                unsigned tensor_d0, unsigned tensor_d1,
                                                unsigned tile_d0, unsigned tile_d1,
                                                unsigned long long stride0_elems) {
    unsigned long long ga = (unsigned long long)(uintptr_t)gaddr;
    u32x4 g0;
    g0[0] = 1u;                                               // count=1, user flags 0
    g0[1] = lds_off;                                          // lds_addr (bytes)
    g0[2] = (unsigned)(ga & 0xFFFFFFFFu);                     // global_addr[31:0]
    g0[3] = (unsigned)((ga >> 32) & 0x1FFFFFFu) | (2u << 30); // addr[56:32] | type=2
    i32x8 g1;
    g1[0] = (int)(2u << 16);                                  // wg_mask=0, data_size=2 (4B)
    g1[1] = (int)((tensor_d0 & 0xFFFFu) << 16);               // tensor_dim0[15:0]
    g1[2] = (int)((tensor_d0 >> 16) | ((tensor_d1 & 0xFFFFu) << 16));
    g1[3] = (int)((tensor_d1 >> 16) | (tile_d0 << 16));       // tile_dim0
    g1[4] = (int)(tile_d1 & 0xFFFFu);                         // tile_dim1, tile_dim2=0
    g1[5] = (int)(unsigned)(stride0_elems & 0xFFFFFFFFu);     // dim0_stride[31:0]
    g1[6] = (int)((stride0_elems >> 32) & 0xFFFFu);
    g1[7] = 0;
    i32x4 gz = {0, 0, 0, 0};
#if __has_include(<hip/amd_detail/amd_gfx1250_TDM.h>)
    i32x8 gz8 = {0, 0, 0, 0, 0, 0, 0, 0};
    __builtin_amdgcn_tensor_load_to_lds(g0, g1, gz, gz, gz8, 0);
#else
    __builtin_amdgcn_tensor_load_to_lds(g0, g1, gz, gz, 0);
#endif
}

// xor-butterfly max within 16-lane rows via v_permlane16_b32 (VALU, not DS)
__device__ __forceinline__ float pmax16(float v, int s0, int s1) {
    int i = __float_as_int(v);
    int p = __builtin_amdgcn_permlane16(i, i, s0, s1, false, false);
    return fmaxf(v, __int_as_float(p));
}

// ---------------------------------------------------------------------------
// Kernel 1: 1x1-conv projections.  x tile arrives via tensor_load_to_lds.
// grid (32, 4), block 128
// ---------------------------------------------------------------------------
__global__ void proj_kernel(const float* __restrict__ x,
                            const float* __restrict__ wq, const float* __restrict__ bq,
                            const float* __restrict__ wk, const float* __restrict__ bk,
                            const float* __restrict__ wv, const float* __restrict__ bv,
                            _Float16* __restrict__ qf, _Float16* __restrict__ kf,
                            _Float16* __restrict__ vt) {
    __shared__ float xs[C_CH][128];
    __shared__ float wqs[CKQ][C_CH], wks[CKQ][C_CH];
    __shared__ float wvs[C_CH][C_CH];
    __shared__ float bqs[CKQ], bks[CKQ], bvs[C_CH];

    const int b    = blockIdx.y;
    const int pos0 = blockIdx.x * 128;
    const int tid  = threadIdx.x;
    const float* xb = x + (size_t)b * C_CH * N_POS;

    // wave 0 kicks the TDM for the whole [64 x 128] x-tile (EXEC ignored by TDM)
    if (tid < 32) {
        unsigned lds_off = (unsigned)(uintptr_t)(&xs[0][0]);
        tdm_load_2d_f32(lds_off, xb + pos0, N_POS, C_CH, 128, C_CH, N_POS);
        __builtin_amdgcn_s_wait_tensorcnt(0);
    }
    // meanwhile every thread pulls weights
    for (int i = tid; i < CKQ * C_CH; i += 128) {
        wqs[i >> 6][i & 63] = wq[i];
        wks[i >> 6][i & 63] = wk[i];
    }
    for (int i = tid; i < C_CH * C_CH; i += 128) wvs[i >> 6][i & 63] = wv[i];
    if (tid < CKQ)  { bqs[tid] = bq[tid]; bks[tid] = bk[tid]; }
    if (tid < C_CH) bvs[tid] = bv[tid];
    __syncthreads();

    const int pos = pos0 + tid;
    float xr[C_CH];
    #pragma unroll
    for (int c = 0; c < C_CH; ++c) xr[c] = xs[c][tid];

    _Float16* qp = qf + ((size_t)b * N_POS + pos) * 32;
    _Float16* kp = kf + ((size_t)b * N_POS + pos) * 32;
    #pragma unroll
    for (int o = 0; o < CKQ; ++o) {
        float qa = bqs[o], ka = bks[o];
        for (int c = 0; c < C_CH; ++c) {
            qa += wqs[o][c] * xr[c];
            ka += wks[o][c] * xr[c];
        }
        qp[o] = (_Float16)qa;
        kp[o] = (_Float16)ka;
    }
    #pragma unroll
    for (int o = CKQ; o < 32; ++o) { qp[o] = (_Float16)0.f; kp[o] = (_Float16)0.f; }

    _Float16* vb = vt + (size_t)b * C_CH * N_POS;
    for (int o = 0; o < C_CH; ++o) {
        float va = bvs[o];
        for (int c = 0; c < C_CH; ++c) va += wvs[o][c] * xr[c];
        vb[(size_t)o * N_POS + pos] = (_Float16)va;
    }
}

// ---------------------------------------------------------------------------
// Kernel 2: flash attention, 64 keys/iter, 14 WMMAs/iter.
// S0/S1 take even/odd keys so each row's (p0,p1)/(p2,p3) pack into single
// b32 LDS stores; storage-column == key, so A-fragment reads and V loads
// stay contiguous.  Loads staged 2 fragments at a time to avoid spills.
// grid (32, 4), block 256 (8 waves, independent 16-query tiles)
// ---------------------------------------------------------------------------
__global__ void attn_kernel(const float* __restrict__ x,
                            const float* __restrict__ gamma,
                            const _Float16* __restrict__ qf,
                            const _Float16* __restrict__ kf,
                            const _Float16* __restrict__ vt,
                            float* __restrict__ out) {
    __shared__ _Float16 pls[8][16][72];   // per-wave 16x64 P staging (+pad)

    const int b     = blockIdx.y;
    const int wave  = threadIdx.x >> 5;
    const int lane  = threadIdx.x & 31;
    const int qrow0 = (blockIdx.x * 8 + wave) * 16;
    const int hl    = lane >> 4;
    const int ln    = lane & 15;
    const int base8 = hl * 8;

    // Q A-fragment (16x32 f16)
    const _Float16* qp = qf + ((size_t)b * N_POS + qrow0 + ln) * 32;
    v8h qlo = *(const v8h*)(qp + hl * 16);
    v8h qhi = *(const v8h*)(qp + hl * 16 + 8);
    v16h aq;
    #pragma unroll
    for (int i = 0; i < 8; ++i) { aq[i] = qlo[i]; aq[i + 8] = qhi[i]; }

    // all-ones B fragment for row-sum WMMA
    v16h ones;
    #pragma unroll
    for (int i = 0; i < 16; ++i) ones[i] = (_Float16)1.0f;

    v8f o0 = {}, o1 = {}, o2 = {}, o3 = {}, osum = {};
    float mrow[8];
    #pragma unroll
    for (int r = 0; r < 8; ++r) mrow[r] = -1e30f;

    const _Float16* kb = kf + (size_t)b * N_POS * 32;
    const _Float16* vb = vt + (size_t)b * C_CH * N_POS;

    for (int k0 = 0; k0 < N_POS; k0 += 64) {
        if (k0 + 64 < N_POS) {
            __builtin_prefetch(kb + (size_t)(k0 + 64 + lane) * 32, 0, 3);
            __builtin_prefetch(vb + (size_t)lane * N_POS + (k0 + 64), 0, 3);
        }

        // S tiles: even keys (2*ln) / odd keys (2*ln+1), staged pairwise
        v8f cz = {};
        v8f s0, s1, s2, s3;
        {
            v16h bk0 = *(const v16h*)(kb + (size_t)(k0 + 2 * ln)     * 32 + hl * 16);
            v16h bk1 = *(const v16h*)(kb + (size_t)(k0 + 2 * ln + 1) * 32 + hl * 16);
            s0 = __builtin_amdgcn_wmma_f32_16x16x32_f16(false, aq, false, bk0, (short)0, cz, false, false);
            s1 = __builtin_amdgcn_wmma_f32_16x16x32_f16(false, aq, false, bk1, (short)0, cz, false, false);
        }
        {
            v16h bk2 = *(const v16h*)(kb + (size_t)(k0 + 32 + 2 * ln)     * 32 + hl * 16);
            v16h bk3 = *(const v16h*)(kb + (size_t)(k0 + 32 + 2 * ln + 1) * 32 + hl * 16);
            s2 = __builtin_amdgcn_wmma_f32_16x16x32_f16(false, aq, false, bk2, (short)0, cz, false, false);
            s3 = __builtin_amdgcn_wmma_f32_16x16x32_f16(false, aq, false, bk3, (short)0, cz, false, false);
        }

        // online softmax: row M = hl*8 + r; storage col == key within tile
        #pragma unroll
        for (int r = 0; r < 8; ++r) {
            float t = fmaxf(fmaxf(s0[r], s1[r]), fmaxf(s2[r], s3[r]));
            t = pmax16(t, (int)0x67452301u, (int)0xEFCDAB89u);  // xor 1
            t = pmax16(t, (int)0x54761032u, (int)0xDCFE98BAu);  // xor 2
            t = pmax16(t, (int)0x32107654u, (int)0xBA98FEDCu);  // xor 4
            t = pmax16(t, (int)0xFEDCBA98u, (int)0x76543210u);  // xor 8
            float mn    = fmaxf(mrow[r], t);
            float alpha = __expf(mrow[r] - mn);
            mrow[r] = mn;
            h2 pa, pb;
            pa[0] = (_Float16)__expf(s0[r] - mn);   // key 2*ln
            pa[1] = (_Float16)__expf(s1[r] - mn);   // key 2*ln+1
            pb[0] = (_Float16)__expf(s2[r] - mn);   // key 32+2*ln
            pb[1] = (_Float16)__expf(s3[r] - mn);   // key 32+2*ln+1
            o0[r] *= alpha; o1[r] *= alpha; o2[r] *= alpha; o3[r] *= alpha;
            osum[r] *= alpha;
            const int mr = hl * 8 + r;
            *(h2*)&pls[wave][mr][     2 * ln] = pa;  // one ds_store_b32
            *(h2*)&pls[wave][mr][32 + 2 * ln] = pb;  // one ds_store_b32
        }
        asm volatile("s_wait_dscnt 0" ::: "memory");

        // P back as two A-fragments (storage cols 0..31 and 32..63)
        const _Float16* pw = &pls[wave][ln][0];
        v8h a0lo = *(const v8h*)(pw + hl * 16);
        v8h a0hi = *(const v8h*)(pw + hl * 16 + 8);
        v8h a1lo = *(const v8h*)(pw + 32 + hl * 16);
        v8h a1hi = *(const v8h*)(pw + 32 + hl * 16 + 8);
        v16h ap0, ap1;
        #pragma unroll
        for (int i = 0; i < 8; ++i) {
            ap0[i] = a0lo[i]; ap0[i + 8] = a0hi[i];
            ap1[i] = a1lo[i]; ap1[i + 8] = a1hi[i];
        }

        // row sums of P via WMMA against ones (replicated across all N)
        osum = __builtin_amdgcn_wmma_f32_16x16x32_f16(false, ap0, false, ones, (short)0, osum, false, false);
        osum = __builtin_amdgcn_wmma_f32_16x16x32_f16(false, ap1, false, ones, (short)0, osum, false, false);

        // PV: stage V fragments pairwise, consume immediately (low pressure)
        {
            v16h bva = *(const v16h*)(vb + (size_t)( 0 + ln) * N_POS + k0      + hl * 16);
            v16h bvb = *(const v16h*)(vb + (size_t)( 0 + ln) * N_POS + k0 + 32 + hl * 16);
            o0 = __builtin_amdgcn_wmma_f32_16x16x32_f16(false, ap0, false, bva, (short)0, o0, false, false);
            o0 = __builtin_amdgcn_wmma_f32_16x16x32_f16(false, ap1, false, bvb, (short)0, o0, false, false);
        }
        {
            v16h bva = *(const v16h*)(vb + (size_t)(16 + ln) * N_POS + k0      + hl * 16);
            v16h bvb = *(const v16h*)(vb + (size_t)(16 + ln) * N_POS + k0 + 32 + hl * 16);
            o1 = __builtin_amdgcn_wmma_f32_16x16x32_f16(false, ap0, false, bva, (short)0, o1, false, false);
            o1 = __builtin_amdgcn_wmma_f32_16x16x32_f16(false, ap1, false, bvb, (short)0, o1, false, false);
        }
        {
            v16h bva = *(const v16h*)(vb + (size_t)(32 + ln) * N_POS + k0      + hl * 16);
            v16h bvb = *(const v16h*)(vb + (size_t)(32 + ln) * N_POS + k0 + 32 + hl * 16);
            o2 = __builtin_amdgcn_wmma_f32_16x16x32_f16(false, ap0, false, bva, (short)0, o2, false, false);
            o2 = __builtin_amdgcn_wmma_f32_16x16x32_f16(false, ap1, false, bvb, (short)0, o2, false, false);
        }
        {
            v16h bva = *(const v16h*)(vb + (size_t)(48 + ln) * N_POS + k0      + hl * 16);
            v16h bvb = *(const v16h*)(vb + (size_t)(48 + ln) * N_POS + k0 + 32 + hl * 16);
            o3 = __builtin_amdgcn_wmma_f32_16x16x32_f16(false, ap0, false, bva, (short)0, o3, false, false);
            o3 = __builtin_amdgcn_wmma_f32_16x16x32_f16(false, ap1, false, bvb, (short)0, o3, false, false);
        }
    }

    // epilogue: out[b][ch][pos] = gamma * O / rowsum + x
    const float g = gamma[0];
    const float* xb = x   + (size_t)b * C_CH * N_POS;
    float*       ob = out + (size_t)b * C_CH * N_POS;
    #pragma unroll
    for (int r = 0; r < 8; ++r) {
        const float inv = 1.0f / osum[r];
        const int pos = qrow0 + hl * 8 + r;
        #pragma unroll
        for (int t = 0; t < 4; ++t) {
            float val = (t == 0 ? o0[r] : t == 1 ? o1[r] : t == 2 ? o2[r] : o3[r]) * inv;
            size_t idx = (size_t)(t * 16 + ln) * N_POS + pos;
            ob[idx] = g * val + xb[idx];
        }
    }
}

// ---------------------------------------------------------------------------
extern "C" void kernel_launch(void* const* d_in, const int* in_sizes, int n_in,
                              void* d_out, int out_size, void* d_ws, size_t ws_size,
                              hipStream_t stream) {
    const float* x     = (const float*)d_in[0];
    const float* wq    = (const float*)d_in[1];
    const float* bq    = (const float*)d_in[2];
    const float* wk    = (const float*)d_in[3];
    const float* bk    = (const float*)d_in[4];
    const float* wv    = (const float*)d_in[5];
    const float* bv    = (const float*)d_in[6];
    const float* gamma = (const float*)d_in[7];
    float* out = (float*)d_out;

    // workspace: qf (1MB) | kf (1MB) | vt (2MB), all f16
    _Float16* qf = (_Float16*)d_ws;
    _Float16* kf = qf + (size_t)4 * N_POS * 32;
    _Float16* vt = kf + (size_t)4 * N_POS * 32;

    proj_kernel<<<dim3(32, 4), 128, 0, stream>>>(x, wq, bq, wk, bk, wv, bv, qf, kf, vt);
    attn_kernel<<<dim3(32, 4), 256, 0, stream>>>(x, gamma, qf, kf, vt, out);
}
